// TemporalMambaBlock_22076131902165
// MI455X (gfx1250) — compile-verified, matrix operand order checked by reading, not run on hardware
//
#include <hip/hip_runtime.h>
#include <hip/hip_bf16.h>

typedef __attribute__((ext_vector_type(16))) _Float16 v16h;
typedef __attribute__((ext_vector_type(8)))  _Float16 v8h;
typedef __attribute__((ext_vector_type(4)))  _Float16 v4h;
typedef __attribute__((ext_vector_type(8)))  float    v8f;

#define DMODEL 128
#define DINNER 256
#define DSTATE 16
#define DTRANK 8
#define SEQT   96
#define NSEQ   828            // 4 * 207
#define NTOK   (NSEQ * SEQT)  // 79488, divisible by 16
#define NPADX  48             // W_x cols padded 40 -> 48

// ---------------- device helpers ----------------

__device__ __forceinline__ float silu_f(float x) {
  return x / (1.0f + __expf(-x));
}
__device__ __forceinline__ float softplus_f(float x) {
  return (x > 20.0f) ? x : log1pf(__expf(x));
}

// Generic 16x32 f16 WMMA fragment load from a row-major matrix with leading
// dimension `ld` (elements). Works for A (rows = M) and, with pre-transposed
// N x K weights, for B (rows = N). Per the ISA layout, lane L (row = L&15,
// g = L>>4) holds two contiguous 8-element K-runs: [8g, 8g+8) and
// [8g+16, 8g+24), pair-packed exactly like consecutive memory halves.
// => two global_load_b128 per lane, zero repacking.
__device__ __forceinline__ v16h loadFrag(const _Float16* __restrict__ P, int ld) {
  int lane = threadIdx.x & 31;
  const _Float16* p = P + (size_t)(lane & 15) * ld + ((lane >> 4) << 3);
  v8h lo = *(const v8h*)p;
  v8h hi = *(const v8h*)(p + 16);
  return __builtin_shufflevector(lo, hi, 0, 1, 2, 3, 4, 5, 6, 7,
                                         8, 9, 10, 11, 12, 13, 14, 15);
}

// ---------------- weight converters (fp32 -> f16, transposed to N x K) ----------------

__global__ void cvt_win_t_kernel(const float* __restrict__ W, _Float16* __restrict__ WT) {
  // W_in (128 x 512) -> WinT (512 x 128)
  int i = blockIdx.x * 256 + threadIdx.x;
  if (i < 512 * DMODEL) {
    int n = i / DMODEL, k = i - n * DMODEL;
    WT[i] = (_Float16)W[k * 512 + n];
  }
}

__global__ void cvt_wx_t_kernel(const float* __restrict__ W, _Float16* __restrict__ WT) {
  // W_x (256 x 40) -> WxT (48 x 256), zero-padded rows 40..47
  int i = blockIdx.x * 256 + threadIdx.x;
  if (i < NPADX * DINNER) {
    int n = i / DINNER, k = i - n * DINNER;
    WT[i] = (n < 40) ? (_Float16)W[k * 40 + n] : (_Float16)0.0f;
  }
}

__global__ void cvt_wout_t_kernel(const float* __restrict__ W, _Float16* __restrict__ WT) {
  // W_out (256 x 128) -> WoutT (128 x 256)
  int i = blockIdx.x * 256 + threadIdx.x;
  if (i < DMODEL * DINNER) {
    int n = i / DINNER, k = i - n * DINNER;
    WT[i] = (_Float16)W[k * DMODEL + n];
  }
}

// ---------------- layernorm: one wave per token ----------------

__global__ void layernorm_kernel(const float* __restrict__ x,
                                 const float* __restrict__ ln_g,
                                 const float* __restrict__ ln_b,
                                 _Float16* __restrict__ hbuf) {
  int wave = threadIdx.x >> 5;
  int lane = threadIdx.x & 31;
  int r = blockIdx.x * 4 + wave;           // token id in (seq-major) order
  int s = r / SEQT, t = r - s * SEQT;
  int b = s / 207, n = s - b * 207;
  const float* row = x + ((((size_t)b * SEQT + t) * 207 + n) * DMODEL);
  const float4 v = ((const float4*)row)[lane];
  float sum = v.x + v.y + v.z + v.w;
#pragma unroll
  for (int o = 16; o > 0; o >>= 1) sum += __shfl_xor(sum, o, 32);
  float mu = sum * (1.0f / 128.0f);
  float d0 = v.x - mu, d1 = v.y - mu, d2 = v.z - mu, d3 = v.w - mu;
  float sq = d0*d0 + d1*d1 + d2*d2 + d3*d3;
#pragma unroll
  for (int o = 16; o > 0; o >>= 1) sq += __shfl_xor(sq, o, 32);
  float rstd = rsqrtf(sq * (1.0f / 128.0f) + 1e-5f);
  int c = lane * 4;
  v4h hv;
  hv[0] = (_Float16)(d0 * rstd * ln_g[c + 0] + ln_b[c + 0]);
  hv[1] = (_Float16)(d1 * rstd * ln_g[c + 1] + ln_b[c + 1]);
  hv[2] = (_Float16)(d2 * rstd * ln_g[c + 2] + ln_b[c + 2]);
  hv[3] = (_Float16)(d3 * rstd * ln_g[c + 3] + ln_b[c + 3]);
  *(v4h*)(hbuf + (size_t)r * DMODEL + c) = hv;
}

// ---------------- GEMM 1: xz = h @ W_in  (K=128, N=512) ----------------
// block = 4 waves, one M-tile per block; wave w covers cols [w*128, w*128+128)

__global__ void gemm_xz_kernel(const _Float16* __restrict__ hbuf,
                               const _Float16* __restrict__ WinT,   // 512 x 128
                               _Float16* __restrict__ ubuf,
                               _Float16* __restrict__ zbuf) {
  int wave = threadIdx.x >> 5;
  int lane = threadIdx.x & 31;
  int tm = blockIdx.x;
  const _Float16* Abase = hbuf + (size_t)tm * 16 * DMODEL;
  int n0 = wave * 128;
  v8f zero = {};
  v8f acc[8];
#pragma unroll
  for (int j = 0; j < 8; ++j) acc[j] = zero;
#pragma unroll
  for (int kc = 0; kc < DMODEL; kc += 32) {
    v16h a = loadFrag(Abase + kc, DMODEL);
#pragma unroll
    for (int j = 0; j < 8; ++j) {
      v16h b = loadFrag(WinT + (size_t)(n0 + j * 16) * DMODEL + kc, DMODEL);
      acc[j] = __builtin_amdgcn_wmma_f32_16x16x32_f16(false, a, false, b, (short)0, acc[j], false, false);
    }
  }
  int grp = lane >> 4;
  int nl  = lane & 15;
#pragma unroll
  for (int j = 0; j < 8; ++j) {
    int n = n0 + j * 16 + nl;
#pragma unroll
    for (int i = 0; i < 8; ++i) {
      int m = tm * 16 + i + 8 * grp;
      float val = acc[j][i];
      if (n < DINNER) ubuf[(size_t)m * DINNER + n] = (_Float16)val;
      else            zbuf[(size_t)m * DINNER + (n - DINNER)] = (_Float16)val;
    }
  }
}

// ---------------- depthwise causal conv(4) + SiLU ----------------

__global__ void conv_silu_kernel(const _Float16* __restrict__ ubuf,
                                 const float* __restrict__ conv_w,
                                 const float* __restrict__ conv_b,
                                 float* __restrict__ ucf,
                                 _Float16* __restrict__ uc16) {
  int c = threadIdx.x;
  int r = blockIdx.x;
  int s = r / SEQT, t = r - s * SEQT;
  float acc = conv_b[c];
#pragma unroll
  for (int k = 0; k < 4; ++k) {
    int tt = t + k - 3;
    if (tt >= 0)
      acc += (float)ubuf[((size_t)s * SEQT + tt) * DINNER + c] * conv_w[c * 4 + k];
  }
  float v = silu_f(acc);
  ucf[(size_t)r * DINNER + c] = v;
  uc16[(size_t)r * DINNER + c] = (_Float16)v;
}

// ---------------- GEMM 2: dbc = uc @ W_x  (K=256, N=48 padded) ----------------
// block = 4 waves, each wave owns its own M-tile (3 accumulators for N=48)

__global__ void gemm_dbc_kernel(const _Float16* __restrict__ uc16,
                                const _Float16* __restrict__ WxT,   // 48 x 256
                                float* __restrict__ dbc) {
  int wave = threadIdx.x >> 5;
  int lane = threadIdx.x & 31;
  int tm = blockIdx.x * 4 + wave;
  const _Float16* Abase = uc16 + (size_t)tm * 16 * DINNER;
  v8f zero = {};
  v8f a0 = zero, a1 = zero, a2 = zero;
#pragma unroll
  for (int kc = 0; kc < DINNER; kc += 32) {
    v16h a  = loadFrag(Abase + kc, DINNER);
    v16h b0 = loadFrag(WxT + (size_t)( 0) * DINNER + kc, DINNER);
    v16h b1 = loadFrag(WxT + (size_t)(16) * DINNER + kc, DINNER);
    v16h b2 = loadFrag(WxT + (size_t)(32) * DINNER + kc, DINNER);
    a0 = __builtin_amdgcn_wmma_f32_16x16x32_f16(false, a, false, b0, (short)0, a0, false, false);
    a1 = __builtin_amdgcn_wmma_f32_16x16x32_f16(false, a, false, b1, (short)0, a1, false, false);
    a2 = __builtin_amdgcn_wmma_f32_16x16x32_f16(false, a, false, b2, (short)0, a2, false, false);
  }
  int grp = lane >> 4;
  int nl  = lane & 15;
#pragma unroll
  for (int i = 0; i < 8; ++i) {
    int m = tm * 16 + i + 8 * grp;
    float* row = dbc + (size_t)m * NPADX;
    row[ 0 + nl] = a0[i];
    row[16 + nl] = a1[i];
    row[32 + nl] = a2[i];
  }
}

// ---------------- selective scan: block = sequence, thread = channel ----------------

__global__ void scan_kernel(const float* __restrict__ dbc,
                            const float* __restrict__ ucf,
                            const float* __restrict__ W_dt,
                            const float* __restrict__ b_dt,
                            const float* __restrict__ A_log,
                            float* __restrict__ ybuf) {
  __shared__ float srow[NPADX];
  int c = threadIdx.x;
  int s = blockIdx.x;
  float A[DSTATE];
#pragma unroll
  for (int j = 0; j < DSTATE; ++j) A[j] = -__expf(A_log[c * DSTATE + j]);
  float wdt[DTRANK];
#pragma unroll
  for (int k = 0; k < DTRANK; ++k) wdt[k] = W_dt[k * DINNER + c];
  float bd = b_dt[c];
  float h[DSTATE];
#pragma unroll
  for (int j = 0; j < DSTATE; ++j) h[j] = 0.0f;

  for (int t = 0; t < SEQT; ++t) {
    size_t r = (size_t)s * SEQT + t;
    __syncthreads();
    if (c < NPADX) srow[c] = dbc[r * NPADX + c];
    __syncthreads();
    float acc = bd;
#pragma unroll
    for (int k = 0; k < DTRANK; ++k) acc += srow[k] * wdt[k];
    float dt = softplus_f(acc);
    float du = dt * ucf[r * DINNER + c];
    float y = 0.0f;
#pragma unroll
    for (int j = 0; j < DSTATE; ++j) {
      float dA = __expf(dt * A[j]);
      h[j] = dA * h[j] + du * srow[DTRANK + j];
      y += h[j] * srow[DTRANK + DSTATE + j];
    }
    ybuf[r * DINNER + c] = y;
  }
}

// ---------------- gating epilogue: v = (y + uc*D) * silu(z) ----------------

__global__ void gate_kernel(const float* __restrict__ ybuf,
                            const float* __restrict__ ucf,
                            const _Float16* __restrict__ zbuf,
                            const float* __restrict__ D_p,
                            _Float16* __restrict__ vbuf) {
  size_t i = (size_t)blockIdx.x * 256 + threadIdx.x;
  int c = (int)(i & 255);
  float z = (float)zbuf[i];
  float v = (ybuf[i] + ucf[i] * D_p[c]) * silu_f(z);
  vbuf[i] = (_Float16)v;
}

// ---------------- GEMM 3: out = v @ W_out + residual  (K=256, N=128) ----------------
// block = 4 waves, one M-tile; wave w covers cols [w*32, w*32+32)

__global__ void gemm_out_kernel(const _Float16* __restrict__ vbuf,
                                const _Float16* __restrict__ WoutT,  // 128 x 256
                                const float* __restrict__ x,
                                float* __restrict__ out) {
  int wave = threadIdx.x >> 5;
  int lane = threadIdx.x & 31;
  int tm = blockIdx.x;
  const _Float16* Abase = vbuf + (size_t)tm * 16 * DINNER;
  int n0 = wave * 32;
  v8f zero = {};
  v8f acc0 = zero, acc1 = zero;
#pragma unroll
  for (int kc = 0; kc < DINNER; kc += 32) {
    v16h a  = loadFrag(Abase + kc, DINNER);
    v16h b0 = loadFrag(WoutT + (size_t)(n0     ) * DINNER + kc, DINNER);
    v16h b1 = loadFrag(WoutT + (size_t)(n0 + 16) * DINNER + kc, DINNER);
    acc0 = __builtin_amdgcn_wmma_f32_16x16x32_f16(false, a, false, b0, (short)0, acc0, false, false);
    acc1 = __builtin_amdgcn_wmma_f32_16x16x32_f16(false, a, false, b1, (short)0, acc1, false, false);
  }
  int grp = lane >> 4;
  int nl  = lane & 15;
#pragma unroll
  for (int j = 0; j < 2; ++j) {
    int n = n0 + j * 16 + nl;
#pragma unroll
    for (int i = 0; i < 8; ++i) {
      int m = tm * 16 + i + 8 * grp;            // token id (seq-major)
      int s = m / SEQT, t = m - s * SEQT;
      int b = s / 207, nn = s - b * 207;
      size_t idx = (((size_t)b * SEQT + t) * 207 + nn) * DMODEL + n;
      float val = (j == 0) ? acc0[i] : acc1[i];
      out[idx] = val + x[idx];
    }
  }
}

// ---------------- host launcher ----------------

extern "C" void kernel_launch(void* const* d_in, const int* in_sizes, int n_in,
                              void* d_out, int out_size, void* d_ws, size_t ws_size,
                              hipStream_t stream) {
  const float* x      = (const float*)d_in[0];
  const float* ln_g   = (const float*)d_in[1];
  const float* ln_b   = (const float*)d_in[2];
  const float* W_in   = (const float*)d_in[3];
  const float* conv_w = (const float*)d_in[4];
  const float* conv_b = (const float*)d_in[5];
  const float* W_x    = (const float*)d_in[6];
  const float* W_dt   = (const float*)d_in[7];
  const float* b_dt   = (const float*)d_in[8];
  const float* A_log  = (const float*)d_in[9];
  const float* D_p    = (const float*)d_in[10];
  const float* W_out  = (const float*)d_in[11];
  float* out = (float*)d_out;

  char* ws = (char*)d_ws;
  size_t off = 0;
  auto alloc = [&](size_t bytes) -> void* {
    void* p = ws + off;
    off += (bytes + 255) & ~(size_t)255;
    return p;
  };

  _Float16* hbuf   = (_Float16*)alloc((size_t)NTOK * DMODEL * 2);
  _Float16* WinT   = (_Float16*)alloc((size_t)512 * DMODEL * 2);
  _Float16* WxT    = (_Float16*)alloc((size_t)NPADX * DINNER * 2);
  _Float16* WoutT  = (_Float16*)alloc((size_t)DMODEL * DINNER * 2);
  _Float16* ubuf   = (_Float16*)alloc((size_t)NTOK * DINNER * 2);
  _Float16* zbuf   = (_Float16*)alloc((size_t)NTOK * DINNER * 2);
  float*    ucf    = (float*)   alloc((size_t)NTOK * DINNER * 4);
  _Float16* uc16   = (_Float16*)alloc((size_t)NTOK * DINNER * 2);
  float*    dbc    = (float*)   alloc((size_t)NTOK * NPADX * 4);
  float*    ybuf   = (float*)   alloc((size_t)NTOK * DINNER * 4);
  _Float16* vbuf   = (_Float16*)alloc((size_t)NTOK * DINNER * 2);

  cvt_win_t_kernel <<<(512 * DMODEL + 255) / 256, 256, 0, stream>>>(W_in, WinT);
  cvt_wx_t_kernel  <<<(NPADX * DINNER + 255) / 256, 256, 0, stream>>>(W_x, WxT);
  cvt_wout_t_kernel<<<(DMODEL * DINNER + 255) / 256, 256, 0, stream>>>(W_out, WoutT);

  layernorm_kernel<<<NTOK / 4, 128, 0, stream>>>(x, ln_g, ln_b, hbuf);
  gemm_xz_kernel  <<<NTOK / 16, 128, 0, stream>>>(hbuf, WinT, ubuf, zbuf);
  conv_silu_kernel<<<NTOK, 256, 0, stream>>>(ubuf, conv_w, conv_b, ucf, uc16);
  gemm_dbc_kernel <<<NTOK / 64, 128, 0, stream>>>(uc16, WxT, dbc);
  scan_kernel     <<<NSEQ, 256, 0, stream>>>(dbc, ucf, W_dt, b_dt, A_log, ybuf);
  gate_kernel     <<<NTOK, 256, 0, stream>>>(ybuf, ucf, zbuf, D_p, vbuf);
  gemm_out_kernel <<<NTOK / 16, 128, 0, stream>>>(vbuf, WoutT, x, out);

  (void)in_sizes; (void)n_in; (void)out_size; (void)ws_size;
}
